// BTTuncker_FC_50955491999868
// MI455X (gfx1250) — matrix operand order repaired and checked by reading, not compiled, not run on hardware
//
#include <hip/hip_runtime.h>
#include <hip/hip_bf16.h>

// ---------------------------------------------------------------------------
// Block-Term Tucker FC for MI455X (gfx1250), wave32 + v_wmma_f32_16x16x32_bf16
//
//   Stage 1: U[(blk,q,jA),cd] = sum_ab P01[(blk,q,jA),ab] * Xn[ab,cd]   (K=64)
//   Stage 2: out[jA,jB]       = sum_kk U[jA,kk] * P23G[kk,jB]          (K=1024)
// with q=(r0,r1), P01 = F0*F1, P23G = sum_{r2,r3} F2*F3*core.
// ---------------------------------------------------------------------------

typedef __attribute__((ext_vector_type(16))) __bf16 v16bf;
typedef __attribute__((ext_vector_type(8)))  float  v8f;
typedef int gv4i __attribute__((vector_size(16)));   // matches async-LDS builtin param

union ABFrag { unsigned int u[8]; v16bf v; };

#if defined(__has_builtin)
#if __has_builtin(__builtin_amdgcn_global_load_async_to_lds_b128)
#define USE_ASYNC_LDS 1
#endif
#endif

__device__ __forceinline__ unsigned short f2bf(float f) {
  unsigned int u = __builtin_bit_cast(unsigned int, f);
  u += 0x7FFFu + ((u >> 16) & 1u);            // round-to-nearest-even
  return (unsigned short)(u >> 16);
}

// 16-bit A-matrix 16x32 K-pair base for VGPR v, lane-half h (ISA 7.12.2)
__device__ __forceinline__ int a_kpair(int v, int h) {
  return (v < 4) ? (8 * h + 2 * v) : (8 + 8 * h + 2 * v);
}

#ifdef USE_ASYNC_LDS
__device__ __forceinline__ void wait_async0() {
#if __has_builtin(__builtin_amdgcn_s_wait_asynccnt)
  __builtin_amdgcn_s_wait_asynccnt(0);
#else
  asm volatile("s_wait_asynccnt 0x0" ::: "memory");
#endif
}
#endif

// ---------------------------------------------------------------------------
// Kernel A: build fused bf16 weights in workspace.
//   wsA1 [m=((blk*4+q)*64+jA)][ab=64]      (65536 bf16)
//   wsB2 [jB=64][kk=((blk*4+q)*64+cd)=1024](65536 bf16)
// ---------------------------------------------------------------------------
__global__ void btt_precompute(const float* __restrict__ cores,
                               const float* __restrict__ factors,
                               unsigned short* __restrict__ wsA1,
                               unsigned short* __restrict__ wsB2) {
  const int tid = threadIdx.x;
  for (int e = tid; e < 65536; e += 256) {
    int m = e >> 6, ab = e & 63;
    int blk = m >> 8, q = (m >> 6) & 3, jA = m & 63;
    int r0 = q >> 1, r1 = q & 1;
    int j0 = jA >> 3, j1 = jA & 7;
    int a = ab >> 3, b2 = ab & 7;
    float f0 = factors[(((blk * 4 + 0) * 8 + a)  * 8 + j0) * 2 + r0];
    float f1 = factors[(((blk * 4 + 1) * 8 + b2) * 8 + j1) * 2 + r1];
    wsA1[e] = f2bf(f0 * f1);
  }
  for (int e = tid; e < 65536; e += 256) {
    int jB = e >> 10, kk = e & 1023;
    int blk = kk >> 8, q = (kk >> 6) & 3, cd = kk & 63;
    int r0 = q >> 1, r1 = q & 1;
    int c = cd >> 3, d = cd & 7;
    int j2 = jB >> 3, j3 = jB & 7;
    float s = 0.f;
#pragma unroll
    for (int r2 = 0; r2 < 2; ++r2)
#pragma unroll
      for (int r3 = 0; r3 < 2; ++r3) {
        float f2v = factors[(((blk * 4 + 2) * 8 + c) * 8 + j2) * 2 + r2];
        float f3v = factors[(((blk * 4 + 3) * 8 + d) * 8 + j3) * 2 + r3];
        float cv  = cores[blk * 16 + r3 * 8 + r2 * 4 + r1 * 2 + r0];
        s += f2v * f3v * cv;
      }
    wsB2[e] = f2bf(s);
  }
}

// ---------------------------------------------------------------------------
// Kernel B: one workgroup (8 wave32) per sample.
// Dynamic LDS layout (297 KB):
//   [0      , 147456): W  — stage1: P01 rows padded to 72 bf16 (1024x72)
//                           stage2: P23G rows padded to 1032 bf16 (64x1032)
//   [147456 , 294912): U  — 1024 rows x 72 bf16 (64 data + 8 pad)
//   [294912 , 304128): XT — Xn transposed, 64 rows x 72 bf16
// ---------------------------------------------------------------------------
__global__ void btt_main(const float* __restrict__ X,
                         const unsigned short* __restrict__ wsA1,
                         const unsigned short* __restrict__ wsB2,
                         float* __restrict__ out) {
  extern __shared__ char smem[];
  unsigned short* Wl = (unsigned short*)smem;
  unsigned short* Ul = (unsigned short*)(smem + 147456);
  unsigned short* Xt = (unsigned short*)(smem + 294912);
  unsigned int* Wu = (unsigned int*)Wl;
  unsigned int* Uu = (unsigned int*)Ul;
  unsigned int* Xu = (unsigned int*)Xt;

  const int tid  = threadIdx.x;
  const int lane = tid & 31;
  const int wave = tid >> 5;
  const int l15  = lane & 15;
  const int h    = lane >> 4;
  const int n    = blockIdx.x;

  __builtin_prefetch(wsA1, 0, 3);   // gfx1250 global_prefetch_b8 (L2-resident weights)
  __builtin_prefetch(wsB2, 0, 3);

  // ---- P01 -> LDS W region (row padding 64 -> 72 bf16) ----
#ifdef USE_ASYNC_LDS
  {
    // 128 KB in 16 B chunks: 8 chunks per 128 B row; DMA directly to LDS
    const char* src = (const char*)wsA1;
    for (int c = tid; c < 8192; c += 256) {
      int row = c >> 3, off = (c & 7) * 16;
      __builtin_amdgcn_global_load_async_to_lds_b128(
          (gv4i*)(src + c * 16),
          (gv4i*)(smem + row * 144 + off),
          0, 0);
    }
  }
#else
  {
    const unsigned int* A1u = (const unsigned int*)wsA1;
    for (int e2 = tid; e2 < 32768; e2 += 256) {
      int row = e2 >> 5, colp = e2 & 31;
      Wu[row * 36 + colp] = A1u[e2];
    }
  }
#endif

  // ---- Load Xn (f32, coalesced float4) -> bf16, transposed XT[cd][ab] ----
  {
    const float4* Xr = (const float4*)(X + (size_t)n * 4096);
    int ab  = tid >> 2;
    int cd0 = (tid & 3) * 16;
#pragma unroll
    for (int j = 0; j < 4; ++j) {
      float4 f = Xr[tid * 4 + j];
      int cd = cd0 + j * 4;
      Xt[(cd + 0) * 72 + ab] = f2bf(f.x);
      Xt[(cd + 1) * 72 + ab] = f2bf(f.y);
      Xt[(cd + 2) * 72 + ab] = f2bf(f.z);
      Xt[(cd + 3) * 72 + ab] = f2bf(f.w);
    }
  }
#ifdef USE_ASYNC_LDS
  wait_async0();
#endif
  __syncthreads();

  // ---- Stage 1: U(1024x64) = P01(1024x64-K) @ Xn(64x64), K=64 ----
  // Wave owns 8 M-tiles; per M-tile the A fragment is shared by 4 N-tiles
  // (4 independent accumulator chains to hide WMMA->WMMA RAW latency).
  for (int mi = 0; mi < 8; ++mi) {
    int m0 = (wave * 8 + mi) * 16;
    v8f acc[4] = {{}, {}, {}, {}};
    int arow = (m0 + l15) * 36;          // uint row base, lane row m
#pragma unroll
    for (int k0 = 0; k0 < 64; k0 += 32) {
      ABFrag a;
#pragma unroll
      for (int v = 0; v < 8; ++v)
        a.u[v] = Wu[arow + (k0 >> 1) + 8 * h + v];
#pragma unroll
      for (int nt = 0; nt < 4; ++nt) {
        ABFrag b;
        int brow = (nt * 16 + l15) * 36; // uint row base, lane col n (XT)
#pragma unroll
        for (int v = 0; v < 8; ++v)
          b.u[v] = Xu[brow + (k0 >> 1) + 8 * h + v];
        acc[nt] = __builtin_amdgcn_wmma_f32_16x16x32_bf16(
            false, a.v, false, b.v, (short)0, acc[nt], false, false);
      }
    }
#pragma unroll
    for (int nt = 0; nt < 4; ++nt)
#pragma unroll
      for (int v = 0; v < 8; ++v)        // D row = v + 8h, col = l15
        Ul[(m0 + 8 * h + v) * 72 + nt * 16 + l15] = f2bf(acc[nt][v]);
  }
  __syncthreads();

  // ---- Recycle W region: P23G rows padded 1024 -> 1032 bf16 ----
#ifdef USE_ASYNC_LDS
  {
    // 128 KB in 16 B chunks: 128 chunks per 2048 B row
    const char* src = (const char*)wsB2;
    for (int c = tid; c < 8192; c += 256) {
      int row = c >> 7, off = (c & 127) * 16;
      __builtin_amdgcn_global_load_async_to_lds_b128(
          (gv4i*)(src + c * 16),
          (gv4i*)(smem + row * 2064 + off),
          0, 0);
    }
    wait_async0();
  }
#else
  {
    const unsigned int* B2u = (const unsigned int*)wsB2;
    for (int e2 = tid; e2 < 32768; e2 += 256) {
      int row = e2 >> 9, colp = e2 & 511;
      Wu[row * 516 + colp] = B2u[e2];
    }
  }
#endif
  __syncthreads();

  // ---- Stage 2: out(64x64) = U(64x1024-K) @ P23G(1024x64), K=1024 ----
  // Each wave: 2 N-tiles sharing one M-tile -> shared A fragment, 2 acc chains.
  {
    int t0  = wave * 2;
    int m0  = (t0 >> 2) * 16;                 // jA tile (same for both)
    int n0a = (t0 & 3) * 16;                  // jB tile 0
    int n0b = ((t0 + 1) & 3) * 16;            // jB tile 1
    v8f acc0 = {}, acc1 = {};
    int browA = (n0a + l15) * 516;
    int browB = (n0b + l15) * 516;
    for (int kb = 0; kb < 1024; kb += 32) {
      ABFrag a, b0, b1;
      int abase = (((kb >> 6) * 64) + m0 + l15) * 36 + ((kb & 32) >> 1);
#pragma unroll
      for (int v = 0; v < 8; ++v) {
        a.u[v]  = Uu[abase + 8 * h + v];
        b0.u[v] = Wu[browA + (kb >> 1) + 8 * h + v];
        b1.u[v] = Wu[browB + (kb >> 1) + 8 * h + v];
      }
      acc0 = __builtin_amdgcn_wmma_f32_16x16x32_bf16(
          false, a.v, false, b0.v, (short)0, acc0, false, false);
      acc1 = __builtin_amdgcn_wmma_f32_16x16x32_bf16(
          false, a.v, false, b1.v, (short)0, acc1, false, false);
    }
    float* orow = out + (size_t)n * 4096;     // out[n, jA*64 + jB]
#pragma unroll
    for (int v = 0; v < 8; ++v) {
      int jA = m0 + 8 * h + v;
      orow[jA * 64 + n0a + l15] = acc0[v];
      orow[jA * 64 + n0b + l15] = acc1[v];
    }
  }
}

extern "C" void kernel_launch(void* const* d_in, const int* in_sizes, int n_in,
                              void* d_out, int out_size, void* d_ws, size_t ws_size,
                              hipStream_t stream) {
  const float* X       = (const float*)d_in[0];
  const float* cores   = (const float*)d_in[1];
  const float* factors = (const float*)d_in[2];
  unsigned short* wsA1 = (unsigned short*)d_ws;       // 65536 bf16 = 128 KB
  unsigned short* wsB2 = wsA1 + 65536;                // 65536 bf16 = 128 KB

  const int nB = in_sizes[0] / 4096;                  // batch (1024)

  btt_precompute<<<1, 256, 0, stream>>>(cores, factors, wsA1, wsB2);
  btt_main<<<nB, 256, 304128, stream>>>(X, wsA1, wsB2, (float*)d_out);
}
